// EmbeddingReverseLayer_66735201845508
// MI455X (gfx1250) — matrix operand order repaired and checked by reading, not compiled, not run on hardware
//
#include <hip/hip_runtime.h>

// ---------------------------------------------------------------------------
// Problem constants (from reference): B=8, S=128, V=50257, E=128
// ---------------------------------------------------------------------------
constexpr int kE      = 128;
constexpr int kV      = 50257;
constexpr int kRows   = 1024;               // B*S
constexpr int kMTiles = kRows / 16;         // 64
constexpr int kNTiles = (kV + 15) / 16;     // 3142
constexpr int kVPad   = kNTiles * 16;       // 50272
constexpr int kKB     = kE / 32;            // 4 K-blocks of 32

// Workspace layout (bytes). Staged arrays hold data in WMMA register order:
// one v8u (8 dwords = 16 bf16) per (tile, kblock, lane).
constexpr size_t SZ_A    = (size_t)kMTiles * kKB * 32 * 8 * 4;   // 256 KiB
constexpr size_t SZ_B    = (size_t)kNTiles * kKB * 32 * 8 * 4;   // ~12.3 MiB
constexpr size_t OFF_AH  = 0;
constexpr size_t OFF_AL  = OFF_AH + SZ_A;
constexpr size_t OFF_BH  = OFF_AL + SZ_A;
constexpr size_t OFF_BL  = OFF_BH + SZ_B;
constexpr size_t OFF_ESQ = OFF_BL + SZ_B;
constexpr size_t OFF_BEST= OFF_ESQ + (size_t)kVPad * 4;          // 1024 x u64

typedef __attribute__((ext_vector_type(8)))  unsigned int v8u;
typedef __attribute__((ext_vector_type(8)))  float        v8f;
typedef __attribute__((ext_vector_type(16))) __bf16       v16bf;

union BfCast { v8u u; v16bf b; };
static __device__ __forceinline__ v16bf as_bf(v8u u) { BfCast c; c.u = u; return c.b; }

// fp32 -> bf16 with round-to-nearest-even (inputs are finite)
static __device__ __forceinline__ unsigned short f2bf(float f) {
    unsigned u = __float_as_uint(f);
    unsigned r = u + 0x7FFFu + ((u >> 16) & 1u);
    return (unsigned short)(r >> 16);
}
static __device__ __forceinline__ float bf2f(unsigned short b) {
    return __uint_as_float(((unsigned)b) << 16);
}

// Monotonic float -> uint mapping (larger float => larger uint)
static __device__ __forceinline__ unsigned f2ord(float f) {
    unsigned u = __float_as_uint(f);
    return (u & 0x80000000u) ? ~u : (u | 0x80000000u);
}

// Running max of packed (score, ~col): equal scores pick the smaller column
static __device__ __forceinline__ void upd(unsigned long long& b, float s, int col) {
    unsigned long long p = ((unsigned long long)f2ord(s) << 32) | (unsigned)(~(unsigned)col);
    if (p > b) b = p;
}

static __device__ __forceinline__ unsigned long long shflx64(unsigned long long v, int m) {
    unsigned lo = (unsigned)v, hi = (unsigned)(v >> 32);
    lo = (unsigned)__shfl_xor((int)lo, m, 32);
    hi = (unsigned)__shfl_xor((int)hi, m, 32);
    return ((unsigned long long)hi << 32) | lo;
}

// ---------------------------------------------------------------------------
// Stage x into WMMA A-matrix (16x32 bf16) register layout, hi/lo split.
// ISA layout: lane L holds row M=L%16, half h=L/16; dword d<4 -> K pair
// (8h+2d, 8h+2d+1); d>=4 -> K pair (16+8h+2(d-4), +1), plus kb*32.
// Also zero-inits the per-row best array (needed every launch).
// ---------------------------------------------------------------------------
__global__ void prep_x_kernel(const float* __restrict__ x, unsigned char* __restrict__ ws) {
    unsigned id = blockIdx.x * blockDim.x + threadIdx.x;
    unsigned long long* best = (unsigned long long*)(ws + OFF_BEST);
    if (id < kRows) best[id] = 0ull;
    if (id >= (unsigned)(kMTiles * kKB * 32 * 8)) return;

    int d = id & 7, lane = (id >> 3) & 31, kb = (id >> 8) & 3, mt = id >> 10;
    int m = mt * 16 + (lane & 15);
    int h = lane >> 4;
    int k = kb * 32 + ((d < 4) ? (8 * h + 2 * d) : (16 + 8 * h + 2 * (d - 4)));

    float f0 = x[m * kE + k], f1 = x[m * kE + k + 1];
    unsigned short h0 = f2bf(f0), h1 = f2bf(f1);
    unsigned short l0 = f2bf(f0 - bf2f(h0)), l1 = f2bf(f1 - bf2f(h1));

    ((unsigned*)(ws + OFF_AH))[id] = ((unsigned)h1 << 16) | h0;
    ((unsigned*)(ws + OFF_AL))[id] = ((unsigned)l1 << 16) | l0;
}

// ---------------------------------------------------------------------------
// Stage embeddings into WMMA B-matrix (32x16 bf16) register layout, hi/lo.
// Lane L holds column N=L%16, half h=L/16; dword d<4 -> K pair
// (8h+2d, +1); d>=4 -> K pair (16+8h+2(d-4), +1), plus kb*32. Pad columns -> 0.
// ---------------------------------------------------------------------------
__global__ void prep_e_kernel(const float* __restrict__ e, unsigned char* __restrict__ ws) {
    unsigned id = blockIdx.x * blockDim.x + threadIdx.x;
    if (id >= (unsigned)(kNTiles * kKB * 32 * 8)) return;

    int d = id & 7, lane = (id >> 3) & 31, kb = (id >> 8) & 3, nt = id >> 10;
    int n = nt * 16 + (lane & 15);
    int h = lane >> 4;
    int k = kb * 32 + ((d < 4) ? (8 * h + 2 * d) : (16 + 8 * h + 2 * (d - 4)));

    float f0 = 0.f, f1 = 0.f;
    if (n < kV) { f0 = e[n * kE + k]; f1 = e[n * kE + k + 1]; }
    unsigned short h0 = f2bf(f0), h1 = f2bf(f1);
    unsigned short l0 = f2bf(f0 - bf2f(h0)), l1 = f2bf(f1 - bf2f(h1));

    ((unsigned*)(ws + OFF_BH))[id] = ((unsigned)h1 << 16) | h0;
    ((unsigned*)(ws + OFF_BL))[id] = ((unsigned)l1 << 16) | l0;
}

// Per-column squared norm (exact fp32); padded columns get +1e30.
__global__ void esq_kernel(const float* __restrict__ e, unsigned char* __restrict__ ws) {
    int v = blockIdx.x * blockDim.x + threadIdx.x;
    if (v >= kVPad) return;
    float* esq = (float*)(ws + OFF_ESQ);
    if (v >= kV) { esq[v] = 1e30f; return; }
    const float4* row = (const float4*)(e + (size_t)v * kE);
    float s = 0.f;
#pragma unroll
    for (int i = 0; i < kE / 4; ++i) {
        float4 q = row[i];
        s += q.x * q.x + q.y * q.y + q.z * q.z + q.w * q.w;
    }
    esq[v] = s;
}

// ---------------------------------------------------------------------------
// Fused GEMM + per-row argmin(dist) = argmax(2*cross - e_sq).
// One wave per block. blockIdx.y = M-pair (32 rows), blockIdx.x = chunk of
// 8 N-pairs (256 columns). hi/lo bf16 split: 3 WMMAs per (tile-pair, kblock).
// C layout: VGPR r, lanes 0-15 -> row r, col=lane; lanes 16-31 -> row r+8.
// ---------------------------------------------------------------------------
#define WMMA_BF16(A, B, C) \
    __builtin_amdgcn_wmma_f32_16x16x32_bf16(false, (A), false, (B), (short)0, (C), false, false)

__launch_bounds__(32)
__global__ void gemm_argmax_kernel(unsigned char* __restrict__ ws) {
    const v8u* Ah = (const v8u*)(ws + OFF_AH);
    const v8u* Al = (const v8u*)(ws + OFF_AL);
    const v8u* Bh = (const v8u*)(ws + OFF_BH);
    const v8u* Bl = (const v8u*)(ws + OFF_BL);
    const float* esq = (const float*)(ws + OFF_ESQ);
    unsigned long long* best = (unsigned long long*)(ws + OFF_BEST);

    const int lane  = threadIdx.x;
    const int mpair = blockIdx.y;              // 0..31  -> M tiles 2*mpair, 2*mpair+1
    const int nc    = blockIdx.x;              // 0..196 -> 8 N-pairs each
    const int npEndAll = kNTiles / 2;          // 1571
    int npBeg = nc * 8;
    int npEnd = npBeg + 8 < npEndAll ? npBeg + 8 : npEndAll;

    unsigned long long bestr[2][8];
#pragma unroll
    for (int mi = 0; mi < 2; ++mi)
#pragma unroll
        for (int r = 0; r < 8; ++r) bestr[mi][r] = 0ull;

    for (int np = npBeg; np < npEnd; ++np) {
        const int nt0 = np * 2;
        v8f acc00 = {}, acc01 = {}, acc10 = {}, acc11 = {};
#pragma unroll
        for (int kb = 0; kb < kKB; ++kb) {
            const int ai0 = ((2 * mpair + 0) * kKB + kb) * 32 + lane;
            const int ai1 = ((2 * mpair + 1) * kKB + kb) * 32 + lane;
            const int bi0 = ((nt0 + 0) * kKB + kb) * 32 + lane;
            const int bi1 = ((nt0 + 1) * kKB + kb) * 32 + lane;
            v16bf a0h = as_bf(Ah[ai0]), a0l = as_bf(Al[ai0]);
            v16bf a1h = as_bf(Ah[ai1]), a1l = as_bf(Al[ai1]);
            v16bf b0h = as_bf(Bh[bi0]), b0l = as_bf(Bl[bi0]);
            v16bf b1h = as_bf(Bh[bi1]), b1l = as_bf(Bl[bi1]);

            acc00 = WMMA_BF16(a0h, b0h, acc00);
            acc00 = WMMA_BF16(a0l, b0h, acc00);
            acc00 = WMMA_BF16(a0h, b0l, acc00);

            acc01 = WMMA_BF16(a0h, b1h, acc01);
            acc01 = WMMA_BF16(a0l, b1h, acc01);
            acc01 = WMMA_BF16(a0h, b1l, acc01);

            acc10 = WMMA_BF16(a1h, b0h, acc10);
            acc10 = WMMA_BF16(a1l, b0h, acc10);
            acc10 = WMMA_BF16(a1h, b0l, acc10);

            acc11 = WMMA_BF16(a1h, b1h, acc11);
            acc11 = WMMA_BF16(a1l, b1h, acc11);
            acc11 = WMMA_BF16(a1h, b1l, acc11);
        }
        const int c0 = nt0 * 16 + (lane & 15);
        const int c1 = c0 + 16;
        const float e0 = esq[c0];
        const float e1 = esq[c1];
#pragma unroll
        for (int r = 0; r < 8; ++r) {
            upd(bestr[0][r], 2.f * acc00[r] - e0, c0);
            upd(bestr[0][r], 2.f * acc01[r] - e1, c1);
            upd(bestr[1][r], 2.f * acc10[r] - e0, c0);
            upd(bestr[1][r], 2.f * acc11[r] - e1, c1);
        }
    }

    // Reduce across the 16 lanes that share each C-matrix row (per half-wave).
#pragma unroll
    for (int mi = 0; mi < 2; ++mi)
#pragma unroll
        for (int r = 0; r < 8; ++r) {
            unsigned long long v = bestr[mi][r];
#pragma unroll
            for (int off = 8; off > 0; off >>= 1) {
                unsigned long long o = shflx64(v, off);
                if (o > v) v = o;
            }
            bestr[mi][r] = v;
        }

    if ((lane & 15) == 0) {
        const int half = lane >> 4;  // lane 0 -> rows [0,8), lane 16 -> rows [8,16)
        for (int mi = 0; mi < 2; ++mi) {
            const int rowBase = (2 * mpair + mi) * 16 + half * 8;
            for (int r = 0; r < 8; ++r)
                atomicMax(&best[rowBase + r], bestr[mi][r]);
        }
    }
}

__global__ void finalize_kernel(const unsigned char* __restrict__ ws, float* __restrict__ out) {
    int i = blockIdx.x * blockDim.x + threadIdx.x;
    if (i >= kRows) return;
    const unsigned long long* best = (const unsigned long long*)(ws + OFF_BEST);
    unsigned idx = ~(unsigned)(best[i] & 0xFFFFFFFFull);
    out[i] = (float)idx;
}

extern "C" void kernel_launch(void* const* d_in, const int* in_sizes, int n_in,
                              void* d_out, int out_size, void* d_ws, size_t ws_size,
                              hipStream_t stream) {
    (void)in_sizes; (void)n_in; (void)out_size; (void)ws_size;
    const float* x   = (const float*)d_in[0];   // [8,128,128] fp32
    const float* emb = (const float*)d_in[1];   // [50257,128] fp32
    float* out       = (float*)d_out;           // [1024] indices as float
    unsigned char* ws = (unsigned char*)d_ws;   // needs ~25.3 MiB

    // Stage A (x) + zero best array
    prep_x_kernel<<<(kMTiles * kKB * 32 * 8 + 255) / 256, 256, 0, stream>>>(x, ws);
    // Stage B (embeddings)
    prep_e_kernel<<<(kNTiles * kKB * 32 * 8 + 255) / 256, 256, 0, stream>>>(emb, ws);
    // Column norms
    esq_kernel<<<(kVPad + 255) / 256, 256, 0, stream>>>(emb, ws);
    // Fused WMMA GEMM + argmax reduction: grid (ceil(1571/8)=197, 32 M-pairs)
    gemm_argmax_kernel<<<dim3(197, 32), 32, 0, stream>>>(ws);
    // Emit indices
    finalize_kernel<<<(kRows + 255) / 256, 256, 0, stream>>>(ws, out);
}
